// GraphAttentionLayer_5660766896650
// MI455X (gfx1250) — matrix operand order repaired and checked by reading, not compiled, not run on hardware
//
#include <hip/hip_runtime.h>
#include <hip/hip_fp16.h>

typedef __attribute__((ext_vector_type(8)))  float    v8f;
typedef __attribute__((ext_vector_type(8)))  _Float16 v8h;
typedef __attribute__((ext_vector_type(16))) _Float16 v16h;

#define B_    8
#define N_    2048
#define FIN   128
#define FOUT  64
#define ALPHA 0.2f
#define NEG_INF (-9.0e15f)

#if __has_builtin(__builtin_amdgcn_global_load_async_to_lds_b128) && \
    __has_builtin(__builtin_amdgcn_s_wait_asynccnt)
#define HAVE_ASYNC_LDS 1
// b128 payload type per the builtin's prototype: vector of 4 ints
typedef int v4i_ __attribute__((vector_size(4 * sizeof(int))));
typedef v4i_ __attribute__((address_space(1))) as1_v4i;
typedef v4i_ __attribute__((address_space(3))) as3_v4i;
__device__ __forceinline__ as1_v4i* to_as1(const void* p) {
  return reinterpret_cast<as1_v4i*>(reinterpret_cast<uintptr_t>(p));
}
__device__ __forceinline__ as3_v4i* to_as3(void* p) {
  // low 32 bits of a flat shared-aperture address are the LDS offset
  return reinterpret_cast<as3_v4i*>((uint32_t)reinterpret_cast<uintptr_t>(p));
}
#endif

// ---------------------------------------------------------------------------
// Kernel 0: swizzle W (128x64 f32) into f16 in B-fragment order so kernel 1
// reads each 32x16 B-fragment as one contiguous 32-byte load.
// Wswz[((kc*2 + hsel)*64 + c)*16 + e] = W[(kc*32 + hsel*16 + e)*64 + c]
// ---------------------------------------------------------------------------
__global__ __launch_bounds__(256) void gat_wswz_kernel(
    const float* __restrict__ W, _Float16* __restrict__ Wswz)
{
  const int idx = blockIdx.x * 256 + threadIdx.x;   // 0..8191
  const int e  = idx & 15;
  const int c  = (idx >> 4) & 63;
  const int hs = (idx >> 10) & 1;
  const int kc = idx >> 11;
  Wswz[idx] = (_Float16)W[(size_t)(kc * 32 + hs * 16 + e) * FOUT + c];
}

// ---------------------------------------------------------------------------
// Kernel 1: Wh = h @ W via v_wmma_f32_16x16x32_f16; also s1 = Wh·a1,
// s2 = Wh·a2 (fp32), and WhT (f16, layout [b][f][j]) for the PV matmul.
// One wave per 16-row tile: grid = 8*128 = 1024 blocks of 32 threads.
// ---------------------------------------------------------------------------
__global__ __launch_bounds__(32) void gat_proj_kernel(
    const float* __restrict__ h, const _Float16* __restrict__ Wswz,
    const float* __restrict__ avec,
    _Float16* __restrict__ WhT, float* __restrict__ s1g, float* __restrict__ s2g)
{
  __shared__ float tile[16 * FOUT];

  const int tid  = blockIdx.x;           // 0..1023
  const int b    = tid >> 7;
  const int i0   = (tid & 127) << 4;
  const int lane = threadIdx.x;          // wave32
  const int n    = lane & 15;            // N-col / A-row within tile
  const int hsel = lane >> 4;            // half-wave select
  const int h8   = hsel * 8;

  const float* hrow = h + ((size_t)b * N_ + i0 + n) * FIN;

  v8f acc[4];
  #pragma unroll
  for (int ft = 0; ft < 4; ++ft) acc[ft] = {};

  #pragma unroll
  for (int kc = 0; kc < FIN / 32; ++kc) {
    // A fragment: 16x32 f16. Lane holds row m=n, K = {hsel*16..hsel*16+15}
    const float* hp = hrow + kc * 32 + hsel * 16;
    const float4 f0 = *(const float4*)(hp);
    const float4 f1 = *(const float4*)(hp + 4);
    const float4 f2 = *(const float4*)(hp + 8);
    const float4 f3 = *(const float4*)(hp + 12);
    v16h af;
    af[0]  = (_Float16)f0.x; af[1]  = (_Float16)f0.y;
    af[2]  = (_Float16)f0.z; af[3]  = (_Float16)f0.w;
    af[4]  = (_Float16)f1.x; af[5]  = (_Float16)f1.y;
    af[6]  = (_Float16)f1.z; af[7]  = (_Float16)f1.w;
    af[8]  = (_Float16)f2.x; af[9]  = (_Float16)f2.y;
    af[10] = (_Float16)f2.z; af[11] = (_Float16)f2.w;
    af[12] = (_Float16)f3.x; af[13] = (_Float16)f3.y;
    af[14] = (_Float16)f3.z; af[15] = (_Float16)f3.w;

    #pragma unroll
    for (int ft = 0; ft < 4; ++ft) {
      // B fragment: one contiguous 32B load from pre-swizzled W
      const v16h bf = *(const v16h*)(
          Wswz + (size_t)(((kc * 2 + hsel) * 64) + ft * 16 + n) * 16);
      acc[ft] = __builtin_amdgcn_wmma_f32_16x16x32_f16(
          false, af, false, bf, (short)0, acc[ft], false, false);
    }
  }

  // Scatter C tiles (f32) to LDS: C[r] = Wh[m = h8 + r][ft*16 + n]
  #pragma unroll
  for (int ft = 0; ft < 4; ++ft)
    #pragma unroll
    for (int r = 0; r < 8; ++r)
      tile[(h8 + r) * FOUT + ft * 16 + n] = acc[ft][r];
  __syncthreads();

  // s1/s2 rows (fp32 precision kept here)
  if (lane < 16) {
    float s1v = 0.f, s2v = 0.f;
    #pragma unroll 8
    for (int f = 0; f < FOUT; ++f) {
      const float w = tile[lane * FOUT + f];
      s1v += w * avec[f];
      s2v += w * avec[FOUT + f];
    }
    s1g[(size_t)b * N_ + i0 + lane] = s1v;
    s2g[(size_t)b * N_ + i0 + lane] = s2v;
  }

  // WhT f16: [b][f][j] so kernel2's B-frag is 16 contiguous halfs
  for (int idx = lane; idx < 16 * FOUT; idx += 32) {
    const int f = idx >> 4, ii = idx & 15;
    WhT[((size_t)b * FOUT + f) * N_ + i0 + ii] = (_Float16)tile[ii * FOUT + f];
  }
}

// ---------------------------------------------------------------------------
// Kernel 2: fused masked-softmax + attention write + P@Wh (WMMA) + ELU.
// One block (128 threads = 4 waves) per 16-row tile. Two passes over adj:
//   A) online softmax statistics (running max, rescaled sum)
//   B) normalized p -> attention output + f16 p-tile (double-buffered LDS,
//      one barrier per 64-j chunk) -> WMMA accumulate
// ---------------------------------------------------------------------------
__global__ __launch_bounds__(128) void gat_attn_kernel(
    const int* __restrict__ adj, const _Float16* __restrict__ WhT,
    const float* __restrict__ s1g, const float* __restrict__ s2g,
    float* __restrict__ out0, float* __restrict__ att)
{
  __shared__ __align__(16) float s2sh[N_];
  __shared__ float s1sh[16];
  __shared__ float rmax[16];
  __shared__ float rinv[16];
  __shared__ __align__(32) _Float16 ptile[2][16 * 64];

  const int blk = blockIdx.x;            // 0..1023
  const int b   = blk >> 7;
  const int i0  = (blk & 127) << 4;
  const int t   = threadIdx.x;

#ifdef HAVE_ASYNC_LDS
  {
    // gfx1250 async copy: global -> LDS without touching VGPRs (ASYNCcnt)
    const float* src = s2g + (size_t)b * N_ + t * 4;
    #pragma unroll
    for (int sweep = 0; sweep < 4; ++sweep) {
      __builtin_amdgcn_global_load_async_to_lds_b128(
          to_as1(src + sweep * 512),
          to_as3(&s2sh[t * 4 + sweep * 512]),
          0, 0);
    }
    __builtin_amdgcn_s_wait_asynccnt(0);
  }
#else
  for (int j = t; j < N_; j += 128) s2sh[j] = s2g[(size_t)b * N_ + j];
#endif
  if (t < 16) s1sh[t] = s1g[(size_t)b * N_ + i0 + t];
  __syncthreads();

  const int   i    = t >> 3;             // row 0..15 (8 threads per row)
  const int   l8   = t & 7;
  const float s1v  = s1sh[i];
  const int*  adjrow = adj + ((size_t)b * N_ + i0 + i) * N_;

  // ---- Pass A: online softmax (max + rescaled sum) over masked row ----
  float m = -__builtin_inff();
  float s = 0.f;
  for (int kk = 0; kk < 32; ++kk) {
    const int j0 = kk * 64 + l8 * 8;
    if (kk + 1 < 32) __builtin_prefetch(adjrow + j0 + 64, 0, 1);
    const int4 a0 = *(const int4*)(adjrow + j0);
    const int4 a1 = *(const int4*)(adjrow + j0 + 4);
    #pragma unroll
    for (int q = 0; q < 8; ++q) {
      const int   av = (q < 4) ? ((const int*)&a0)[q] : ((const int*)&a1)[q - 4];
      const float x  = s1v + s2sh[j0 + q];
      const float e  = x > 0.f ? x : ALPHA * x;
      const float v  = av > 0 ? e : NEG_INF;
      const float mn = fmaxf(m, v);                 // branchless online update
      s = s * __expf(m - mn) + __expf(v - mn);
      m = mn;
    }
  }
  // butterfly-merge the 8 partials of each row (wave32, width-8 groups)
  #pragma unroll
  for (int off = 1; off < 8; off <<= 1) {
    const float om = __shfl_xor(m, off, 8);
    const float os = __shfl_xor(s, off, 8);
    const float M  = fmaxf(m, om);
    s = s * __expf(m - M) + os * __expf(om - M);
    m = M;
  }
  if (l8 == 0) { rmax[i] = m; rinv[i] = 1.0f / s; }
  __syncthreads();
  const float rm = rmax[i];
  const float ri = rinv[i];

  // ---- Pass B: normalized p -> attention + WMMA accumulate ----
  const int wave = t >> 5;               // 4 waves -> 4 F-tiles of 16
  const int lane = t & 31;
  const int n    = lane & 15;
  const int hsel = lane >> 4;
  const int h8   = hsel * 8;

  // B-frag base: WhT[b][wave*16+n][hsel*16 + e]  (16 contiguous halfs)
  const _Float16* bbase =
      WhT + ((size_t)b * FOUT + wave * 16 + n) * N_ + hsel * 16;
  float* attrow = att + ((size_t)b * N_ + i0 + i) * N_;

  v8f acc = {};
  for (int jc = 0; jc < 32; ++jc) {
    _Float16* buf = ptile[jc & 1];
    const int j0 = jc * 64 + l8 * 8;
    const int4 a0 = *(const int4*)(adjrow + j0);
    const int4 a1 = *(const int4*)(adjrow + j0 + 4);
    float4 pv0, pv1;
    v8h ph;
    #pragma unroll
    for (int q = 0; q < 8; ++q) {
      const int   av = (q < 4) ? ((const int*)&a0)[q] : ((const int*)&a1)[q - 4];
      const float x  = s1v + s2sh[j0 + q];
      const float e  = x > 0.f ? x : ALPHA * x;
      const float v  = av > 0 ? e : NEG_INF;
      const float p  = __expf(v - rm) * ri;
      if (q < 4) ((float*)&pv0)[q] = p; else ((float*)&pv1)[q - 4] = p;
      ph[q] = (_Float16)p;
    }
    *(float4*)(attrow + j0)     = pv0;   // normalized attention out
    *(float4*)(attrow + j0 + 4) = pv1;
    *(v8h*)&buf[i * 64 + l8 * 8] = ph;   // f16 p-tile, A-frag friendly
    __syncthreads();                     // single barrier (double-buffered)

    #pragma unroll
    for (int u = 0; u < 2; ++u) {
      // A frag: row m=n, one contiguous 16-half run at hsel*16 (matches B)
      const v8h lo = *(const v8h*)&buf[n * 64 + u * 32 + hsel * 16];
      const v8h hi = *(const v8h*)&buf[n * 64 + u * 32 + hsel * 16 + 8];
      v16h af;
      #pragma unroll
      for (int q = 0; q < 8; ++q) { af[q] = lo[q]; af[q + 8] = hi[q]; }
      const v16h bf = *(const v16h*)(bbase + jc * 64 + u * 32);
      acc = __builtin_amdgcn_wmma_f32_16x16x32_f16(
          false, af, false, bf, (short)0, acc, false, false);
    }
  }

  // Epilogue: ELU + store h_out tile. C[r] = H[h8+r][wave*16+n]
  #pragma unroll
  for (int r = 0; r < 8; ++r) {
    const int   mrow = h8 + r;
    const float v    = acc[r];
    const float o    = v > 0.f ? v : (__expf(v) - 1.0f);
    out0[((size_t)b * N_ + i0 + mrow) * FOUT + wave * 16 + n] = o;
  }
}

// ---------------------------------------------------------------------------
extern "C" void kernel_launch(void* const* d_in, const int* in_sizes, int n_in,
                              void* d_out, int out_size, void* d_ws, size_t ws_size,
                              hipStream_t stream) {
  (void)in_sizes; (void)n_in; (void)out_size; (void)ws_size;
  const float* h    = (const float*)d_in[0];   // (8,2048,128) f32
  const int*   adj  = (const int*)d_in[1];     // (8,2048,2048) i32
  const float* W    = (const float*)d_in[2];   // (128,64) f32
  const float* avec = (const float*)d_in[3];   // (128,1) f32

  // workspace: WhT f16 (2 MB) | s1 f32 (64 KB) | s2 f32 (64 KB) | Wswz (16 KB)
  _Float16* WhT  = (_Float16*)d_ws;
  float*    s1g  = (float*)((char*)d_ws + (size_t)B_ * FOUT * N_ * sizeof(_Float16));
  float*    s2g  = s1g + (size_t)B_ * N_;
  _Float16* Wswz = (_Float16*)(s2g + (size_t)B_ * N_);

  float* out0 = (float*)d_out;                         // elu(h_out): 8*2048*64
  float* att  = out0 + (size_t)B_ * N_ * FOUT;         // attention:  8*2048*2048

  gat_wswz_kernel<<<(FIN * FOUT) / 256, 256, 0, stream>>>(W, Wswz);
  gat_proj_kernel<<<B_ * (N_ / 16), 32, 0, stream>>>(h, Wswz, avec, WhT, s1g, s2g);
  gat_attn_kernel<<<B_ * (N_ / 16), 128, 0, stream>>>(adj, WhT, s1g, s2g, out0, att);
}
